// EnhancedTimeAwareFDR_75788992905554
// MI455X (gfx1250) — matrix operand order repaired and checked by reading, not compiled, not run on hardware
//
#include <hip/hip_runtime.h>

#define NN 32768
#define DD 128

typedef __attribute__((ext_vector_type(16))) _Float16 v16h;
typedef __attribute__((ext_vector_type(8)))  _Float16 v8h;
typedef __attribute__((ext_vector_type(8)))  float    v8f;

__device__ __forceinline__ float sigm(float x){ return 1.0f/(1.0f+expf(-x)); }

__device__ __forceinline__ v8f wmma16(v16h a, v16h b, v8f c){
  return __builtin_amdgcn_wmma_f32_16x16x32_f16(false, a, false, b, (short)0, c, false, false);
}

// A-fragment (16x32 f16, row-major source): lane L row L%16 holds K
// {base+hi*16 .. +15} contiguously in halves 0-15.
__device__ __forceinline__ v16h load_a_frag(const _Float16* p){
  v8h lo = *(const v8h*)p;
  v8h hi = *(const v8h*)(p + 8);
  v16h a;
#pragma unroll
  for (int i = 0; i < 8; ++i){ a[i] = lo[i]; a[i+8] = hi[i]; }
  return a;
}

// ---------------- prep: fp32 -> fp16 weight/operand copies ----------------
__global__ void k_prep16(const float* wih, const float* whh, const float* iq,
                         _Float16* wih16, _Float16* whh16, _Float16* iq16){
  int i = blockIdx.x*256 + threadIdx.x;
  if (i < 384*128){ wih16[i] = (_Float16)wih[i]; whh16[i] = (_Float16)whh[i]; }
  if (i < 16*128) iq16[i] = (_Float16)iq[i];
}

// ---------------- time feature + layernorm + x -----------------------------
__global__ void k_timefeat(const float* emb, const float* tpos, const float* te_w,
                           const float* te_b, const float* ln_g, const float* ln_b,
                           float* x, _Float16* x16, _Float16* e16){
  int row = blockIdx.x, j = threadIdx.x;            // 128 threads
  float t = tpos[row];
  float v = fmaxf(t*te_w[j*2+0] + te_w[j*2+1] + te_b[j], 0.0f);
  __shared__ float red[128];
  red[j] = v; __syncthreads();
  for (int s = 64; s > 0; s >>= 1){ if (j < s) red[j] += red[j+s]; __syncthreads(); }
  float mu = red[0] * (1.0f/128.0f); __syncthreads();
  float dv = v - mu;
  red[j] = dv*dv; __syncthreads();
  for (int s = 64; s > 0; s >>= 1){ if (j < s) red[j] += red[j+s]; __syncthreads(); }
  float var = red[0] * (1.0f/128.0f);
  float ln = dv * rsqrtf(var + 1e-5f) * ln_g[j] + ln_b[j];
  float e  = emb[row*128 + j];
  float xv = e + ln;
  x  [row*128 + j] = xv;
  x16[row*128 + j] = (_Float16)xv;
  e16[row*128 + j] = (_Float16)e;
}

// ---------------- smoothed scores / vad ------------------------------------
__global__ void k_scores(const float* pred, const float* vad, const float* gk,
                         float* ab, float* vd, float* mix){
  int n = blockIdx.x*256 + threadIdx.x;
  if (n >= NN) return;
  float s[5] = {0,0,0,0,0};
#pragma unroll
  for (int k = 0; k < 5; ++k){
    int nn = n + k - 2;
    if (nn >= 0 && nn < NN){
      float g = gk[k];
#pragma unroll
      for (int c = 0; c < 5; ++c) s[c] += g * pred[nn*5 + c];
    }
  }
  float mx = s[0];
#pragma unroll
  for (int c = 1; c < 5; ++c) mx = fmaxf(mx, s[c]);
  float se = 0.0f;
#pragma unroll
  for (int c = 0; c < 5; ++c){ s[c] = expf(s[c]-mx); se += s[c]; }
  float abv = 1.0f - s[0]/se;
  float vv  = 1.0f/(1.0f + expf(vad[n*2+0] - vad[n*2+1]));
  ab[n] = abv; vd[n] = vv; mix[n] = 0.5f*(abv+vv);
}

// ---------------- two scalar GRUs (irreducibly serial) ---------------------
__global__ void k_gru1(const float* ab, const float* vd,
                       const float* wia, const float* wha, const float* bia, const float* bha,
                       const float* wiv, const float* whv, const float* biv, const float* bhv,
                       float* gagv){
  int tid = threadIdx.x;
  if (tid >= 2) return;
  const float* seq = tid ? vd : ab;
  const float* wi = tid ? wiv : wia;  const float* wh = tid ? whv : wha;
  const float* bi = tid ? biv : bia;  const float* bh = tid ? bhv : bha;
  float wi0=wi[0],wi1=wi[1],wi2=wi[2], wh0=wh[0],wh1=wh[1],wh2=wh[2];
  float bi0=bi[0],bi1=bi[1],bi2=bi[2], bh0=bh[0],bh1=bh[1],bh2=bh[2];
  float h = 0.0f;
  for (int t = 0; t < NN; ++t){
    float xv = seq[t];
    float r = sigm(xv*wi0 + bi0 + h*wh0 + bh0);
    float z = sigm(xv*wi1 + bi1 + h*wh1 + bh1);
    float n = tanhf(xv*wi2 + bi2 + r*(h*wh2 + bh2));
    h = (1.0f - z)*n + z*h;
  }
  gagv[tid] = h;
}

// ---------------- attention logits via WMMA: S = iq @ E^T ------------------
__global__ __launch_bounds__(256) void k_attn_logits(const _Float16* iq16,
                                                     const _Float16* e16, float* S){
  int lane = threadIdx.x & 31, wave = threadIdx.x >> 5;
  int ntile = blockIdx.x*8 + wave;              // 2048 node tiles
  int rlo = lane & 15, hi = lane >> 4;
  v8f acc = {};
#pragma unroll
  for (int kc = 0; kc < 4; ++kc){
    v16h a = load_a_frag(&iq16[rlo*128 + kc*32 + hi*16]);
    v16h b = *(const v16h*)&e16[(ntile*16 + rlo)*128 + kc*32 + hi*16];
    acc = wmma16(a, b, acc);
  }
#pragma unroll
  for (int r = 0; r < 8; ++r)
    S[(r + 8*hi)*NN + ntile*16 + rlo] = acc[r];
}

// ---------------- per-row max + 1/sumexp -----------------------------------
__global__ void k_rowstats(const float* S, float* stat){
  int m = blockIdx.x, tid = threadIdx.x;        // 256 threads
  __shared__ float red[256];
  float mx = -1e30f;
  for (int n = tid; n < NN; n += 256) mx = fmaxf(mx, S[m*NN + n]);
  red[tid] = mx; __syncthreads();
  for (int s = 128; s > 0; s >>= 1){ if (tid < s) red[tid] = fmaxf(red[tid], red[tid+s]); __syncthreads(); }
  mx = red[0]; __syncthreads();
  float sm = 0.0f;
  for (int n = tid; n < NN; n += 256) sm += expf(S[m*NN + n] - mx);
  red[tid] = sm; __syncthreads();
  for (int s = 128; s > 0; s >>= 1){ if (tid < s) red[tid] += red[tid+s]; __syncthreads(); }
  if (tid == 0){ stat[m*2] = mx; stat[m*2+1] = 1.0f/red[0]; }
}

// ---------------- qf = softmax(S) @ x --------------------------------------
__global__ void k_qf(const float* S, const float* stat, const float* x, float* qf){
  int m = blockIdx.x, tid = threadIdx.x;        // 128 threads
  float mx = stat[m*2], inv = stat[m*2+1];
  __shared__ float wsh[128];
  float acc = 0.0f;
  for (int base = 0; base < NN; base += 128){
    if (base + 128 < NN)
      __builtin_prefetch(&x[(size_t)(base + 128)*128 + tid], 0, 1);
    wsh[tid] = expf(S[m*NN + base + tid] - mx) * inv;
    __syncthreads();
#pragma unroll 4
    for (int j = 0; j < 128; ++j) acc += wsh[j] * x[(base + j)*128 + tid];
    __syncthreads();
  }
  qf[m*128 + tid] = acc;
}

// ---------------- interval generator MLP -> starts/ends --------------------
__global__ void k_ig(const float* qf, const float* gagv,
                     const float* w1, const float* b1, const float* w2, const float* b2,
                     float* se){
  __shared__ float q[16*130];
  __shared__ float h1[16*128];
  __shared__ float psh[16*4];
  int tid = threadIdx.x;                        // 128 threads
  for (int i = tid; i < 16*128; i += 128) q[(i>>7)*130 + (i & 127)] = qf[i];
  if (tid < 16){ q[tid*130 + 128] = gagv[0]; q[tid*130 + 129] = gagv[1]; }
  __syncthreads();
  for (int m = 0; m < 16; ++m){
    float a = b1[tid];
    for (int k = 0; k < 130; ++k) a += q[m*130 + k] * w1[tid*130 + k];
    h1[m*128 + tid] = fmaxf(a, 0.0f);
  }
  __syncthreads();
  if (tid < 64){
    int m = tid >> 2, o = tid & 3;
    float a = b2[o];
    for (int k = 0; k < 128; ++k) a += h1[m*128 + k] * w2[o*128 + k];
    psh[m*4 + o] = a;
  }
  __syncthreads();
  if (tid < 16){
    float c = sigm(psh[tid*4 + 0]);
    float w = sigm(psh[tid*4 + 1]) * 0.5f;
    se[tid]      = fminf(fmaxf(c - 0.5f*w, 0.0f), 1.0f);
    se[16 + tid] = fminf(fmaxf(c + 0.5f*w, 0.0f), 1.0f);
  }
}

// ---------------- masks ----------------------------------------------------
__global__ void k_mask(const float* tpos, const float* se, unsigned char* mb){
  int n = blockIdx.x*256 + threadIdx.x;
  if (n >= NN) return;
  float t = tpos[n];
#pragma unroll
  for (int m = 0; m < 16; ++m)
    mb[n*16 + m] = (t >= se[m] && t <= se[16 + m]) ? 1 : 0;
}

// ---------------- gi_all = x @ lf_wih^T + bih via WMMA ---------------------
__global__ __launch_bounds__(256) void k_gi(const _Float16* x16, const _Float16* wih16,
                                            const float* bih, float* gi){
  int lane = threadIdx.x & 31, wave = threadIdx.x >> 5;
  int tau = blockIdx.x*8 + wave;                // 2048 x 24 tiles
  int rt = tau / 24, ct = tau % 24;
  int rlo = lane & 15, hi = lane >> 4;
  v8f acc = {};
#pragma unroll
  for (int kc = 0; kc < 4; ++kc){
    v16h a = load_a_frag(&x16[(rt*16 + rlo)*128 + kc*32 + hi*16]);
    v16h b = *(const v16h*)&wih16[(ct*16 + rlo)*128 + kc*32 + hi*16];
    acc = wmma16(a, b, acc);
  }
  int col = ct*16 + rlo;
  float bv = bih[col];
#pragma unroll
  for (int r = 0; r < 8; ++r)
    gi[(size_t)(rt*16 + r + 8*hi)*384 + col] = acc[r] + bv;
}

// ---------------- the serial local-feature GRU scan (WMMA per step) --------
// gi rows are double-buffered in LDS via async global->LDS DMA (ASYNCcnt),
// issued one step ahead so the HBM/L2 latency is hidden behind the WMMAs.
__global__ __launch_bounds__(256) void k_scan(const _Float16* whh16, const float* bhh,
                                              const float* gi, const unsigned char* mb,
                                              const float* mix,
                                              const float* law, const float* lah,
                                              const float* labi, const float* labh,
                                              float* lfeat, float* lab){
  __shared__ __align__(16) float hf[2048];
  __shared__ __align__(32) _Float16 hf16[2048];
  __shared__ float rbuf[2048];
  __shared__ float zbuf[2048];
  __shared__ float hnbuf[2048];
  __shared__ __align__(16) float gi_s[2*384];   // double buffer: 1536 B each
  __shared__ float ha[16];
  int tid = threadIdx.x, lane = tid & 31, wave = tid >> 5;
  int rlo = lane & 15, hi = lane >> 4;

  for (int i = tid; i < 2048; i += 256){ hf[i] = 0.0f; hf16[i] = (_Float16)0.0f; }
  if (tid < 16) ha[tid] = 0.0f;

  // constant B fragments of lf_whh^T, resident in VGPRs for the whole scan
  v16h Bf[3][4];
  float bhc[3]; int colc[3];
#pragma unroll
  for (int ti = 0; ti < 3; ++ti){
    int tile = wave*3 + ti;
#pragma unroll
    for (int kc = 0; kc < 4; ++kc)
      Bf[ti][kc] = *(const v16h*)&whh16[(tile*16 + rlo)*128 + kc*32 + hi*16];
    int c = tile*16 + rlo;
    colc[ti] = c; bhc[ti] = bhh[c];
  }
  float lw0=law[0], lw1=law[1], lw2=law[2];
  float lh0=lah[0], lh1=lah[1], lh2=lah[2];
  float li0=labi[0], li1=labi[1], li2=labi[2];
  float lb0=labh[0], lb1=labh[1], lb2=labh[2];

  // async DMA of one 384-float gi row into LDS buffer b (wave 0 only).
  // GVS form: per-lane 16B, inst offset applies to BOTH LDS and global sides.
  unsigned long long gbase = (unsigned long long)(uintptr_t)gi;
  unsigned lbase0 = (unsigned)(uintptr_t)(&gi_s[0]) + (unsigned)lane*16u;
  unsigned lbase1 = (unsigned)(uintptr_t)(&gi_s[384]) + (unsigned)lane*16u;
  auto issue_row = [&](int r, int b){
    unsigned va = (unsigned)r*1536u + (unsigned)lane*16u;
    unsigned ld = b ? lbase1 : lbase0;
    asm volatile("global_load_async_to_lds_b128 %0, %1, %2 offset:0"
                 :: "v"(ld), "v"(va), "s"(gbase) : "memory");
    asm volatile("global_load_async_to_lds_b128 %0, %1, %2 offset:512"
                 :: "v"(ld), "v"(va), "s"(gbase) : "memory");
    asm volatile("global_load_async_to_lds_b128 %0, %1, %2 offset:1024"
                 :: "v"(ld), "v"(va), "s"(gbase) : "memory");
  };
  if (wave == 0){
    issue_row(0, 0);
    asm volatile("s_wait_asynccnt 0" ::: "memory");
  }
  __syncthreads();

  for (int t = 0; t < NN; ++t){
    // kick off DMA of row t+1 into the buffer not read this step
    if (wave == 0 && (t + 1) < NN) issue_row(t + 1, (t + 1) & 1);

    v8f a0 = {}, a1 = {}, a2 = {};
#pragma unroll
    for (int kc = 0; kc < 4; ++kc){
      v16h a = load_a_frag(&hf16[rlo*128 + kc*32 + hi*16]);
      a0 = wmma16(a, Bf[0][kc], a0);
      a1 = wmma16(a, Bf[1][kc], a1);
      a2 = wmma16(a, Bf[2][kc], a2);
    }
    const float* girow = &gi_s[(t & 1)*384];
#pragma unroll
    for (int ti = 0; ti < 3; ++ti){
      v8f av = (ti == 0) ? a0 : ((ti == 1) ? a1 : a2);
      int c = colc[ti];
      float giv = girow[c];
      float bv  = bhc[ti];
#pragma unroll
      for (int r = 0; r < 8; ++r){
        int m = r + 8*hi;
        float gh = av[r] + bv;
        if (c < 128)       rbuf [m*128 + c]         = sigm(giv + gh);
        else if (c < 256)  zbuf [m*128 + (c - 128)] = sigm(giv + gh);
        else               hnbuf[m*128 + (c - 256)] = gh;
      }
    }
    __syncthreads();
#pragma unroll
    for (int it = 0; it < 8; ++it){
      int e = tid + it*256;
      int m = e >> 7, d = e & 127;
      float rv = rbuf[e], zv = zbuf[e], hn = hnbuf[e];
      float nv = tanhf(girow[256 + d] + rv*hn);
      float h  = hf[e];
      float nh = mb[t*16 + m] ? (1.0f - zv)*nv + zv*h : h;
      hf[e] = nh; hf16[e] = (_Float16)nh;
    }
    if (tid < 16){
      float at = mix[t], h = ha[tid];
      float r = sigm(at*lw0 + li0 + h*lh0 + lb0);
      float z = sigm(at*lw1 + li1 + h*lh1 + lb1);
      float n = tanhf(at*lw2 + li2 + r*(h*lh2 + lb2));
      ha[tid] = mb[t*16 + tid] ? (1.0f - z)*n + z*h : h;
    }
    // row t+1 must be resident before any wave enters step t+1
    if (wave == 0) asm volatile("s_wait_asynccnt 0" ::: "memory");
    __syncthreads();
  }
  for (int i = tid; i < 2048; i += 256) lfeat[i] = hf[i];
  if (tid < 16) lab[tid] = ha[tid];
}

// ---------------- refinement + all outputs ---------------------------------
__global__ __launch_bounds__(256) void k_final(const float* qf, const float* lfeat,
                                               const float* lab_g, const float* se_g,
                                               const float* alen,
                                               const float* rw1, const float* rb1,
                                               const float* rw2, const float* rb2,
                                               const float* wp, const float* cw, const float* cb,
                                               const float* clw, const float* clb,
                                               float* LG, float* out){
  __shared__ float qn[2048];
  __shared__ float lf[2048];
  __shared__ float red[256];
  __shared__ float la[16], st[16], en[16];
  __shared__ float q[16*133];
  __shared__ float hq[16*256];
  __shared__ float lse2[32];
  int tid = threadIdx.x;

  for (int i = tid; i < 2048; i += 256){ qn[i] = qf[i]; lf[i] = lfeat[i]; }
  if (tid < 16){ la[tid] = lab_g[tid]; st[tid] = se_g[tid]; en[tid] = se_g[16 + tid]; }
  __syncthreads();

  // diversity: G = qn qn^T on unit rows
  if (tid < 16){
    float s = 0.0f;
    for (int d = 0; d < 128; ++d) s += qn[tid*128 + d]*qn[tid*128 + d];
    red[tid] = 1.0f / fmaxf(sqrtf(s), 1e-8f);
  }
  __syncthreads();
  for (int i = tid; i < 2048; i += 256) qn[i] *= red[i >> 7];
  __syncthreads();
  float part = 0.0f;
  { int i = tid >> 4, j = tid & 15;
    if (j > i){ float s = 0.0f; for (int d = 0; d < 128; ++d) s += qn[i*128+d]*qn[j*128+d]; part = s; } }
  red[tid] = part; __syncthreads();
  for (int s = 128; s > 0; s >>= 1){ if (tid < s) red[tid] += red[tid+s]; __syncthreads(); }
  if (tid == 0) out[113] = red[0] / 120.0f;
  __syncthreads();

  // 3 refinement layers
  for (int l = 0; l < 3; ++l){
    for (int i = tid; i < 2048; i += 256){ int m = i >> 7, d = i & 127; q[m*133 + d] = lf[i]; }
    if (tid < 16){
      float s = st[tid], e = en[tid];
      q[tid*133 + 128] = 0.5f*(s + e);
      q[tid*133 + 129] = e - s;
      q[tid*133 + 130] = s;
      q[tid*133 + 131] = e;
      q[tid*133 + 132] = la[tid];
    }
    __syncthreads();
    { const float* w = &rw1[(l*256 + tid)*133];
      float b = rb1[l*256 + tid];
      for (int m = 0; m < 16; ++m){
        float a = b;
        for (int k = 0; k < 133; ++k) a += q[m*133 + k] * w[k];
        hq[m*256 + tid] = fmaxf(a, 0.0f);
      } }
    __syncthreads();
    if (tid < 200){
      const float* w = &rw2[(l*200 + tid)*256];
      float b = rb2[l*200 + tid];
      for (int m = 0; m < 16; ++m){
        float a = b;
        for (int k = 0; k < 256; ++k) a += hq[m*256 + k] * w[k];
        LG[(l*16 + m)*200 + tid] = a;
      }
    }
    __syncthreads();
    if (tid < 32){
      int m = tid >> 1, sd = tid & 1;
      const float* Lr = &LG[(l*16 + m)*200 + sd*100];
      float mx = -1e30f;
      for (int b = 0; b < 100; ++b) mx = fmaxf(mx, Lr[b]);
      float se = 0.0f, dot = 0.0f;
      for (int b = 0; b < 100; ++b){ float e0 = expf(Lr[b]-mx); se += e0; dot += e0*wp[b]; }
      float off = dot / se;
      if (sd == 0) st[m] = fminf(fmaxf(st[m] + off, 0.0f), 1.0f);
      else         en[m] = fminf(fmaxf(en[m] + off, 0.0f), 1.0f);
    }
    __syncthreads();
  }

  if (tid < 16){ float al = alen[0]; out[tid*2] = st[tid]*al; out[tid*2+1] = en[tid]*al; }

  // KL distillation term
  if (tid < 32){
    int m = tid >> 1, sd = tid & 1;
    const float* Lr = &LG[(2*16 + m)*200 + sd*100];
    float mx = -1e30f; for (int b = 0; b < 100; ++b) mx = fmaxf(mx, Lr[b]);
    float s = 0.0f;    for (int b = 0; b < 100; ++b) s += expf(Lr[b]-mx);
    lse2[tid] = mx + logf(s);
  }
  __syncthreads();
  float dpart = 0.0f;
  if (tid < 32){
    int m = tid >> 1, sd = tid & 1;
    const float* L2 = &LG[(2*16 + m)*200 + sd*100];
    float l2 = lse2[tid];
    for (int l = 0; l < 3; ++l){
      const float* Lr = &LG[(l*16 + m)*200 + sd*100];
      float mx = -1e30f; for (int b = 0; b < 100; ++b) mx = fmaxf(mx, Lr[b]);
      float s = 0.0f;    for (int b = 0; b < 100; ++b) s += expf(Lr[b]-mx);
      float lsel = mx + logf(s);
      for (int b = 0; b < 100; ++b){
        float x2 = L2[b] - l2;
        dpart += expf(x2) * (x2 - (Lr[b] - lsel));
      }
    }
    dpart *= 0.01f;
  }
  red[tid] = dpart; __syncthreads();
  for (int s = 128; s > 0; s >>= 1){ if (tid < s) red[tid] += red[tid+s]; __syncthreads(); }
  if (tid == 0) out[32] = red[0];

  // conf / cls heads
  if (tid < 16){
    float a = cb[0];
    for (int k = 0; k < 128; ++k) a += lf[tid*128 + k] * cw[k];
    a += la[tid] * cw[128];
    out[33 + tid] = a;
  }
  if (tid < 64){
    int m = tid >> 2, c = tid & 3;
    const float* w = &clw[c*129];
    float a = clb[c];
    for (int k = 0; k < 128; ++k) a += lf[m*128 + k] * w[k];
    a += la[m] * w[128];
    out[49 + m*4 + c] = a;
  }
}

extern "C" void kernel_launch(void* const* d_in, const int* in_sizes, int n_in,
                              void* d_out, int out_size, void* d_ws, size_t ws_size,
                              hipStream_t stream){
  const float* emb  = (const float*)d_in[0];
  const float* tpos = (const float*)d_in[1];
  const float* pred = (const float*)d_in[2];
  const float* vad  = (const float*)d_in[3];
  const float* alen = (const float*)d_in[4];
  const float* te_w = (const float*)d_in[5];
  const float* te_b = (const float*)d_in[6];
  const float* ln_g = (const float*)d_in[7];
  const float* ln_b = (const float*)d_in[8];
  const float* gk   = (const float*)d_in[9];
  const float* gaw  = (const float*)d_in[10];
  const float* gah  = (const float*)d_in[11];
  const float* gabi = (const float*)d_in[12];
  const float* gabh = (const float*)d_in[13];
  const float* gvw  = (const float*)d_in[14];
  const float* gvh  = (const float*)d_in[15];
  const float* gvbi = (const float*)d_in[16];
  const float* gvbh = (const float*)d_in[17];
  const float* iq   = (const float*)d_in[18];
  const float* igw1 = (const float*)d_in[19];
  const float* igb1 = (const float*)d_in[20];
  const float* igw2 = (const float*)d_in[21];
  const float* igb2 = (const float*)d_in[22];
  const float* wih  = (const float*)d_in[23];
  const float* whh  = (const float*)d_in[24];
  const float* bih  = (const float*)d_in[25];
  const float* bhh  = (const float*)d_in[26];
  const float* law  = (const float*)d_in[27];
  const float* lah  = (const float*)d_in[28];
  const float* labi = (const float*)d_in[29];
  const float* labh = (const float*)d_in[30];
  const float* rw1  = (const float*)d_in[31];
  const float* rb1  = (const float*)d_in[32];
  const float* rw2  = (const float*)d_in[33];
  const float* rb2  = (const float*)d_in[34];
  const float* wp   = (const float*)d_in[35];
  const float* cw   = (const float*)d_in[36];
  const float* cb   = (const float*)d_in[37];
  const float* clw  = (const float*)d_in[38];
  const float* clb  = (const float*)d_in[39];
  (void)in_sizes; (void)n_in; (void)out_size; (void)ws_size;

  char* w = (char*)d_ws;
  size_t off = 0;
  auto bump = [&](size_t bytes)->char*{
    off = (off + 255) & ~(size_t)255;
    char* p = w + off;
    off += bytes;
    return p;
  };
  float*     x      = (float*)    bump((size_t)NN*128*4);
  _Float16*  x16    = (_Float16*) bump((size_t)NN*128*2);
  _Float16*  e16    = (_Float16*) bump((size_t)NN*128*2);
  _Float16*  wih16  = (_Float16*) bump((size_t)384*128*2);
  _Float16*  whh16  = (_Float16*) bump((size_t)384*128*2);
  _Float16*  iq16   = (_Float16*) bump((size_t)16*128*2);
  float*     S      = (float*)    bump((size_t)16*NN*4);
  float*     stat   = (float*)    bump(32*4);
  float*     qf     = (float*)    bump(2048*4);
  float*     gagv   = (float*)    bump(2*4);
  float*     ab     = (float*)    bump((size_t)NN*4);
  float*     vd     = (float*)    bump((size_t)NN*4);
  float*     mix    = (float*)    bump((size_t)NN*4);
  float*     se     = (float*)    bump(32*4);
  unsigned char* mb = (unsigned char*)bump((size_t)NN*16);
  float*     gi     = (float*)    bump((size_t)NN*384*4);
  float*     lfeat  = (float*)    bump(2048*4);
  float*     lab    = (float*)    bump(16*4);
  float*     LG     = (float*)    bump((size_t)3*16*200*4);

  k_prep16     <<<192, 256, 0, stream>>>(wih, whh, iq, wih16, whh16, iq16);
  k_timefeat   <<<NN, 128, 0, stream>>>(emb, tpos, te_w, te_b, ln_g, ln_b, x, x16, e16);
  k_scores     <<<NN/256, 256, 0, stream>>>(pred, vad, gk, ab, vd, mix);
  k_gru1       <<<1, 32, 0, stream>>>(ab, vd, gaw, gah, gabi, gabh, gvw, gvh, gvbi, gvbh, gagv);
  k_attn_logits<<<256, 256, 0, stream>>>(iq16, e16, S);
  k_rowstats   <<<16, 256, 0, stream>>>(S, stat);
  k_qf         <<<16, 128, 0, stream>>>(S, stat, x, qf);
  k_ig         <<<1, 128, 0, stream>>>(qf, gagv, igw1, igb1, igw2, igb2, se);
  k_mask       <<<NN/256, 256, 0, stream>>>(tpos, se, mb);
  k_gi         <<<6144, 256, 0, stream>>>(x16, wih16, bih, gi);
  k_scan       <<<1, 256, 0, stream>>>(whh16, bhh, gi, mb, mix, law, lah, labi, labh, lfeat, lab);
  k_final      <<<1, 256, 0, stream>>>(qf, lfeat, lab, se, alen, rw1, rb1, rw2, rb2,
                                       wp, cw, cb, clw, clb, LG, (float*)d_out);
}
